// SageSLAAttentionImpl_79731772883271
// MI455X (gfx1250) — 16-bit and 8-bit matrix operand order repaired and checked by reading, compiled, not run on hardware
//
#include <hip/hip_runtime.h>
#include <stdint.h>

#define BB 2
#define LL 2048
#define HH 16
#define DD 128
#define BLKQ 128
#define BLKK 64
#define NQB (LL / BLKQ)
#define NKB (LL / BLKK)
#define TOPK 16
#define FP8MAXV (448.0f / 2.25f)

typedef __attribute__((ext_vector_type(8)))  int    v8i;
typedef __attribute__((ext_vector_type(8)))  float  v8f;
typedef __attribute__((ext_vector_type(16))) __bf16 v16bf;

#if __has_builtin(__builtin_amdgcn_tensor_load_to_lds) && \
    __has_builtin(__builtin_amdgcn_s_wait_tensorcnt)
#define USE_TDM 1
#else
#define USE_TDM 0
#endif

__device__ __forceinline__ int kbaseA8(int j, int hi) {
  return (j & 1) * 4 + ((j >> 1) & 1) * 16 + (j >> 2) * 32 + hi * 8;
}
__device__ __forceinline__ int kbaseB8(int j, int hi) {
  return (j & 1) * 4 + ((j >> 1) & 1) * 16 + (j >> 2) * 32 + hi * 8;
}
__device__ __forceinline__ int kmapA16(int i, int hi) {
  int j = i >> 1, t = i & 1;
  return ((j < 4) ? 2 * j : 2 * j + 8) + t + hi * 8;
}
__device__ __forceinline__ int kmapB16(int i, int hi) { return ((i < 8) ? i : i + 8) + hi * 8; }

__device__ __forceinline__ unsigned int f32_to_e4m3(float x) {
  unsigned int u = __float_as_uint(x);
  unsigned int sgn = (u >> 24) & 0x80u;
  float ax = fminf(fabsf(x), 448.0f);
  if (ax < 0.015625f) {
    unsigned int m = (unsigned int)__builtin_rintf(ax * 512.0f);
    return sgn | (m & 0xFu);
  }
  unsigned int b = __float_as_uint(ax) + 0x00080000u;
  int E = (int)((b >> 23) & 0xFF) - 120;
  unsigned int m = (b >> 20) & 7u;
  if (E > 15) { E = 15; m = 6u; }
  if (E == 15 && m == 7u) m = 6u;
  return sgn | ((unsigned)E << 3) | m;
}

#if USE_TDM
typedef __attribute__((ext_vector_type(4))) unsigned int u32x4;
typedef __attribute__((ext_vector_type(8))) int          i32x8;
typedef __attribute__((ext_vector_type(4))) int          i32x4;

__device__ __forceinline__ void tdm_load_2d(const void* gsrc, unsigned lds_off,
                                            unsigned tile_x, unsigned tile_y,
                                            unsigned tensor_x, unsigned tensor_y,
                                            unsigned stride_x) {
  const unsigned long long ga = (unsigned long long)(size_t)gsrc;
  u32x4 g0;
  g0[0] = 1u;
  g0[1] = lds_off;
  g0[2] = (unsigned)(ga & 0xFFFFFFFFu);
  g0[3] = (unsigned)((ga >> 32) & 0x01FFFFFFu) | (2u << 30);
  i32x8 g1;
  g1[0] = 0;
  g1[1] = (int)((tensor_x & 0xFFFFu) << 16);
  g1[2] = (int)(((tensor_x >> 16) & 0xFFFFu) | ((tensor_y & 0xFFFFu) << 16));
  g1[3] = (int)(((tensor_y >> 16) & 0xFFFFu) | ((tile_x & 0xFFFFu) << 16));
  g1[4] = (int)(tile_y & 0xFFFFu);
  g1[5] = (int)stride_x;
  g1[6] = 0;
  g1[7] = 0;
  i32x4 z4 = {0, 0, 0, 0};
#if __clang_major__ >= 23
  i32x8 z8 = {0, 0, 0, 0, 0, 0, 0, 0};
  __builtin_amdgcn_tensor_load_to_lds(g0, g1, z4, z4, z8, 0);
#else
  __builtin_amdgcn_tensor_load_to_lds(g0, g1, z4, z4, 0);
#endif
}
#endif

__global__ void k_prep(const float* __restrict__ Q, const float* __restrict__ K,
                       const float* __restrict__ V, float* km, float* vscale,
                       float* ksum, float* kvf32, float* qp, float* kp) {
  const int h = blockIdx.x, b = blockIdx.y, d = threadIdx.x;
  const int bh = b * HH + h;
  const size_t base = ((size_t)b * LL * HH + h) * DD + d;
  float ks = 0.f, vm = 0.f;
  for (int l = 0; l < LL; ++l) {
    ks += K[base + (size_t)l * HH * DD];
    vm = fmaxf(vm, fabsf(V[base + (size_t)l * HH * DD]));
  }
  km[bh * DD + d] = ks / (float)LL;
  vscale[bh * DD + d] = vm / FP8MAXV + 1e-8f;
  ksum[bh * DD + d] = 0.f;
  for (int i = d; i < DD * DD; i += DD) kvf32[(size_t)bh * DD * DD + i] = 0.f;
  for (int qb = 0; qb < NQB; ++qb) {
    float a = 0.f;
    for (int i = 0; i < BLKQ; ++i) a += Q[base + (size_t)(qb * BLKQ + i) * HH * DD];
    qp[(bh * NQB + qb) * DD + d] = a / (float)BLKQ;
  }
  for (int kb = 0; kb < NKB; ++kb) {
    float a = 0.f;
    for (int i = 0; i < BLKK; ++i) a += K[base + (size_t)(kb * BLKK + i) * HH * DD];
    kp[(bh * NKB + kb) * DD + d] = a / (float)BLKK;
  }
}

__global__ void k_cvtw(const float* __restrict__ pw, __bf16* pwb) {
  int i = blockIdx.x * blockDim.x + threadIdx.x;
  if (i < DD * DD) pwb[i] = (__bf16)pw[i];
}

__global__ void k_topk(const float* __restrict__ qp, const float* __restrict__ kp,
                       unsigned* maskbuf) {
  const int h = blockIdx.x, b = blockIdx.y, tid = threadIdx.x;
  const int bh = b * HH + h;
  __shared__ float sim[NQB * NKB];
  for (int p = tid; p < NQB * NKB; p += blockDim.x) {
    int qb = p / NKB, kb = p % NKB;
    float a = 0.f;
    for (int d = 0; d < DD; ++d)
      a += qp[(bh * NQB + qb) * DD + d] * kp[(bh * NKB + kb) * DD + d];
    sim[p] = a;
  }
  __syncthreads();
  if (tid < NQB) {
    unsigned m = 0;
    for (int t = 0; t < TOPK; ++t) {
      float best = -3.0e38f; int bi = 0;
      for (int j = 0; j < NKB; ++j)
        if (!((m >> j) & 1u) && sim[tid * NKB + j] > best) { best = sim[tid * NKB + j]; bi = j; }
      m |= 1u << bi;
    }
    maskbuf[bh * NQB + tid] = m;
  }
}

__global__ void k_quant_q(const float* __restrict__ Q, signed char* q8, float* qscale) {
  const int qb = blockIdx.x, h = blockIdx.y, b = blockIdx.z, r = threadIdx.x;
  const int bh = b * HH + h;
  const int l = qb * BLKQ + r;
  const float* src = Q + ((size_t)(b * LL + l) * HH + h) * DD;
  float am = 0.f;
  for (int d = 0; d < DD; ++d) am = fmaxf(am, fabsf(src[d]));
  __shared__ float red[BLKQ];
  red[r] = am; __syncthreads();
  for (int s = BLKQ / 2; s > 0; s >>= 1) {
    if (r < s) red[r] = fmaxf(red[r], red[r + s]);
    __syncthreads();
  }
  const float sc = red[0] / 127.f + 1e-8f;
  if (r == 0) qscale[bh * NQB + qb] = sc;
  const float inv = 1.f / sc;
  int* dst = (int*)(q8 + ((size_t)bh * LL + l) * DD);
  for (int d = 0; d < DD; d += 4) {
    int p = 0;
#pragma unroll
    for (int t = 0; t < 4; ++t) {
      float v = fminf(fmaxf(__builtin_rintf(src[d + t] * inv), -127.f), 127.f);
      p |= ((int)v & 0xFF) << (8 * t);
    }
    dst[d >> 2] = p;
  }
}

__global__ void k_quant_k(const float* __restrict__ K, const float* __restrict__ km,
                          signed char* k8, float* kscale) {
  const int kb = blockIdx.x, h = blockIdx.y, b = blockIdx.z, r = threadIdx.x;
  const int bh = b * HH + h;
  const int l = kb * BLKK + r;
  const float* src = K + ((size_t)(b * LL + l) * HH + h) * DD;
  const float* kmr = km + (size_t)bh * DD;
  float am = 0.f;
  for (int d = 0; d < DD; ++d) am = fmaxf(am, fabsf(src[d] - kmr[d]));
  __shared__ float red[BLKK];
  red[r] = am; __syncthreads();
  for (int s = BLKK / 2; s > 0; s >>= 1) {
    if (r < s) red[r] = fmaxf(red[r], red[r + s]);
    __syncthreads();
  }
  const float sc = red[0] / 127.f + 1e-8f;
  if (r == 0) kscale[bh * NKB + kb] = sc;
  const float inv = 1.f / sc;
  int* dst = (int*)(k8 + ((size_t)bh * LL + l) * DD);
  for (int d = 0; d < DD; d += 4) {
    int p = 0;
#pragma unroll
    for (int t = 0; t < 4; ++t) {
      float v = fminf(fmaxf(__builtin_rintf((src[d + t] - kmr[d + t]) * inv), -127.f), 127.f);
      p |= ((int)v & 0xFF) << (8 * t);
    }
    dst[d >> 2] = p;
  }
}

__global__ void k_rowwise(const float* __restrict__ Q, const float* __restrict__ K,
                          const float* __restrict__ V, const float* __restrict__ vscale,
                          float* ksum, __bf16* qfb, __bf16* kftb, __bf16* vbt,
                          unsigned char* v8t) {
  const int h = blockIdx.y, b = blockIdx.z;
  const int warp = threadIdx.x >> 5, lane = threadIdx.x & 31;
  const int l = blockIdx.x * 8 + warp;
  const int bh = b * HH + h;
  const size_t rbase = ((size_t)(b * LL + l) * HH + h) * DD;
  float qv[4], kv[4], vv[4];
#pragma unroll
  for (int j = 0; j < 4; ++j) {
    int d = lane + 32 * j;
    qv[j] = Q[rbase + d]; kv[j] = K[rbase + d]; vv[j] = V[rbase + d];
  }
  float qm = fmaxf(fmaxf(qv[0], qv[1]), fmaxf(qv[2], qv[3]));
  float kmx = fmaxf(fmaxf(kv[0], kv[1]), fmaxf(kv[2], kv[3]));
#pragma unroll
  for (int s = 1; s < 32; s <<= 1) {
    qm = fmaxf(qm, __shfl_xor(qm, s, 32));
    kmx = fmaxf(kmx, __shfl_xor(kmx, s, 32));
  }
  float qsum = 0.f, ksm = 0.f;
#pragma unroll
  for (int j = 0; j < 4; ++j) {
    qv[j] = __expf(qv[j] - qm); kv[j] = __expf(kv[j] - kmx);
    qsum += qv[j]; ksm += kv[j];
  }
#pragma unroll
  for (int s = 1; s < 32; s <<= 1) {
    qsum += __shfl_xor(qsum, s, 32);
    ksm += __shfl_xor(ksm, s, 32);
  }
  const float qi = 1.f / qsum, ki = 1.f / ksm;
#pragma unroll
  for (int j = 0; j < 4; ++j) {
    const int d = lane + 32 * j;
    const float qf = qv[j] * qi, kf = kv[j] * ki;
    qfb[((size_t)bh * LL + l) * DD + d] = (__bf16)qf;
    kftb[((size_t)bh * DD + d) * LL + l] = (__bf16)kf;
    atomicAdd(&ksum[bh * DD + d], kf);
    const float vs = vscale[bh * DD + d];
    v8t[((size_t)bh * DD + d) * LL + l] = (unsigned char)f32_to_e4m3(vv[j] / vs);
    vbt[((size_t)bh * DD + d) * LL + l] = (__bf16)vv[j];
  }
}

__global__ void k_kvgemm(const __bf16* __restrict__ kftb, const __bf16* __restrict__ vbt,
                         float* kvf32) {
  const int kc = blockIdx.x, h = blockIdx.y, b = blockIdx.z;
  const int bh = b * HH + h;
  const int w = threadIdx.x >> 5, lane = threadIdx.x & 31;
  const int lane16 = lane & 15, hi = lane >> 4;
  const int m0 = w * 16;
  v8f acc[8] = {};
  for (int ks = kc * 8; ks < kc * 8 + 8; ++ks) {
    v16bf A;
    v16bf Bfs[8];
#pragma unroll
    for (int i = 0; i < 16; ++i)
      A[i] = kftb[((size_t)bh * DD + m0 + lane16) * LL + ks * 32 + kmapA16(i, hi)];
#pragma unroll
    for (int n = 0; n < 8; ++n)
#pragma unroll
      for (int i = 0; i < 16; ++i)
        Bfs[n][i] = vbt[((size_t)bh * DD + n * 16 + lane16) * LL + ks * 32 + kmapB16(i, hi)];
#pragma unroll
    for (int n = 0; n < 8; ++n)
      acc[n] = __builtin_amdgcn_wmma_f32_16x16x32_bf16(false, A, false, Bfs[n], (short)0,
                                                       acc[n], false, false);
  }
#pragma unroll
  for (int n = 0; n < 8; ++n)
#pragma unroll
    for (int v = 0; v < 8; ++v)
      atomicAdd(&kvf32[((size_t)bh * DD + n * 16 + lane16) * DD + m0 + v + 8 * hi],
                acc[n][v]);
}

__global__ void k_lingemm(const __bf16* __restrict__ qfb, const float* __restrict__ kvf32,
                          const float* __restrict__ ksum, __bf16* olb) {
  const int h = blockIdx.y, b = blockIdx.z;
  const int l0 = blockIdx.x * 128;
  const int bh = b * HH + h;
  const int tid = threadIdx.x;
  __shared__ float den[128];
  {
    float a = 0.f;
    const __bf16* qr = qfb + ((size_t)bh * LL + l0 + tid) * DD;
    for (int d = 0; d < DD; ++d) a += (float)qr[d] * ksum[bh * DD + d];
    den[tid] = 1e-5f + a;
  }
  __syncthreads();
  const int w = tid >> 5, lane = tid & 31, lane16 = lane & 15, hi = lane >> 4;
  const int m0 = w * 32;
  v8f acc[2][8] = {};
  for (int ks = 0; ks < DD / 32; ++ks) {
    v16bf A[2];
    v16bf Bfs[8];
#pragma unroll
    for (int mt = 0; mt < 2; ++mt)
#pragma unroll
      for (int i = 0; i < 16; ++i)
        A[mt][i] = qfb[((size_t)bh * LL + l0 + m0 + mt * 16 + lane16) * DD + ks * 32 +
                       kmapA16(i, hi)];
#pragma unroll
    for (int n = 0; n < 8; ++n)
#pragma unroll
      for (int i = 0; i < 16; ++i)
        Bfs[n][i] = (__bf16)kvf32[((size_t)bh * DD + n * 16 + lane16) * DD + ks * 32 +
                                  kmapB16(i, hi)];
#pragma unroll
    for (int n = 0; n < 8; ++n)
#pragma unroll
      for (int mt = 0; mt < 2; ++mt)
        acc[mt][n] = __builtin_amdgcn_wmma_f32_16x16x32_bf16(false, A[mt], false, Bfs[n],
                                                             (short)0, acc[mt][n], false, false);
  }
#pragma unroll
  for (int mt = 0; mt < 2; ++mt)
#pragma unroll
    for (int n = 0; n < 8; ++n)
#pragma unroll
      for (int v = 0; v < 8; ++v) {
        const int row = m0 + mt * 16 + v + 8 * hi;
        olb[((size_t)bh * LL + l0 + row) * DD + n * 16 + lane16] =
            (__bf16)(acc[mt][n][v] / den[row]);
      }
}

__global__ void k_proj(const __bf16* __restrict__ olb, const __bf16* __restrict__ pwb,
                       const float* __restrict__ projb, float* out) {
  const int h = blockIdx.y, b = blockIdx.z;
  const int l0 = blockIdx.x * 128;
  const int bh = b * HH + h;
  const int tid = threadIdx.x;
  const int w = tid >> 5, lane = tid & 31, lane16 = lane & 15, hi = lane >> 4;
  const int m0 = w * 32;
  v8f acc[2][8] = {};
  for (int ks = 0; ks < DD / 32; ++ks) {
    v16bf A[2];
    v16bf Bfs[8];
#pragma unroll
    for (int mt = 0; mt < 2; ++mt)
#pragma unroll
      for (int i = 0; i < 16; ++i)
        A[mt][i] = olb[((size_t)bh * LL + l0 + m0 + mt * 16 + lane16) * DD + ks * 32 +
                       kmapA16(i, hi)];
#pragma unroll
    for (int n = 0; n < 8; ++n)
#pragma unroll
      for (int i = 0; i < 16; ++i)
        Bfs[n][i] = pwb[(n * 16 + lane16) * DD + ks * 32 + kmapB16(i, hi)];
#pragma unroll
    for (int n = 0; n < 8; ++n)
#pragma unroll
      for (int mt = 0; mt < 2; ++mt)
        acc[mt][n] = __builtin_amdgcn_wmma_f32_16x16x32_bf16(false, A[mt], false, Bfs[n],
                                                             (short)0, acc[mt][n], false, false);
  }
#pragma unroll
  for (int mt = 0; mt < 2; ++mt)
#pragma unroll
    for (int n = 0; n < 8; ++n) {
      const float pb = projb[n * 16 + lane16];
#pragma unroll
      for (int v = 0; v < 8; ++v) {
        const int l = l0 + m0 + mt * 16 + v + 8 * hi;
        out[((size_t)(b * LL + l) * HH + h) * DD + n * 16 + lane16] = acc[mt][n][v] + pb;
      }
    }
}

__global__ void __launch_bounds__(256) k_attn(
    const signed char* __restrict__ q8, const signed char* __restrict__ k8,
    const unsigned char* __restrict__ v8t, const float* __restrict__ qscale,
    const float* __restrict__ kscale, const float* __restrict__ vscale,
    const unsigned* __restrict__ maskbuf, float* out) {
  const int qb = blockIdx.x, h = blockIdx.y, b = blockIdx.z;
  const int bh = b * HH + h;
  const int tid = threadIdx.x, w = tid >> 5, lane = tid & 31;
  const int lane16 = lane & 15, hi = lane >> 4;
  const int m0 = w * 16;
  const float qs = qscale[bh * NQB + qb];
  const float smul = 0.08838834764831845f;

  __shared__ __align__(16) unsigned char kbuf[2][BLKK * DD];
  __shared__ __align__(16) unsigned char vbuf[2][DD * BLKK];
  __shared__ unsigned char p_lds[8 * 16 * 64];
  __shared__ int seln[NKB + 1];
  unsigned char* myp = p_lds + w * (16 * 64);

  if (tid == 0) {
    const unsigned mask = maskbuf[bh * NQB + qb];
    int c = 0;
    for (int kb = 0; kb < NKB; ++kb)
      if ((mask >> kb) & 1u) seln[c++] = kb;
    seln[NKB] = c;
  }
  __syncthreads();
  const int cnt = __builtin_amdgcn_readfirstlane(seln[NKB]);

  auto stage = [&](int kb, int buf) {
#if USE_TDM
    if (tid < 32) {
      tdm_load_2d(k8 + ((size_t)bh * LL + (size_t)kb * BLKK) * DD,
                  (unsigned)(size_t)&kbuf[buf][0], DD, BLKK, DD, LL, DD);
      tdm_load_2d(v8t + (size_t)bh * DD * LL + (size_t)kb * BLKK,
                  (unsigned)(size_t)&vbuf[buf][0], BLKK, DD, LL, DD, LL);
    }
#else
    const int4* ksrc = (const int4*)(k8 + ((size_t)bh * LL + (size_t)kb * BLKK) * DD);
    for (int idx = tid; idx < (BLKK * DD) / 16; idx += 256)
      ((int4*)&kbuf[buf][0])[idx] = ksrc[idx];
    for (int idx = tid; idx < (DD * BLKK) / 16; idx += 256) {
      const int d = idx >> 2, c4 = idx & 3;
      ((int4*)&vbuf[buf][d * BLKK])[c4] =
          *(const int4*)(v8t + ((size_t)bh * DD + d) * LL + (size_t)kb * BLKK + c4 * 16);
    }
#endif
  };

  v8i Aq[2];
#pragma unroll
  for (int kstep = 0; kstep < 2; ++kstep) {
    const signed char* rowp =
        q8 + ((size_t)bh * LL + qb * BLKQ + m0 + lane16) * DD + kstep * 64;
#pragma unroll
    for (int j = 0; j < 8; ++j)
      Aq[kstep][j] = *(const int*)(rowp + kbaseA8(j, hi));
  }

  float mrun[8], lrun[8];
  v8f Oacc[8] = {};
#pragma unroll
  for (int v = 0; v < 8; ++v) { mrun[v] = -1e30f; lrun[v] = 0.f; }

  if (cnt > 0) stage(__builtin_amdgcn_readfirstlane(seln[0]), 0);

  for (int i = 0; i < cnt; ++i) {
    const int buf = i & 1;
    const int kb = __builtin_amdgcn_readfirstlane(seln[i]);
#if USE_TDM
    if (tid < 32) __builtin_amdgcn_s_wait_tensorcnt(0);
#endif
    __syncthreads();
    if (i + 1 < cnt) stage(__builtin_amdgcn_readfirstlane(seln[i + 1]), buf ^ 1);

    const unsigned char* kcur = &kbuf[buf][0];
    const unsigned char* vcur = &vbuf[buf][0];
    const float sfac = qs * kscale[bh * NKB + kb] * smul;

    float sf[4][8];
#pragma unroll
    for (int nt = 0; nt < 4; ++nt) {
      v8i Bk[2];
#pragma unroll
      for (int kstep = 0; kstep < 2; ++kstep)
#pragma unroll
        for (int j = 0; j < 8; ++j)
          Bk[kstep][j] =
              *(const int*)(kcur + (nt * 16 + lane16) * DD + kstep * 64 + kbaseB8(j, hi));
      v8i sacc = {};
      sacc = __builtin_amdgcn_wmma_i32_16x16x64_iu8(true, Aq[0], true, Bk[0], sacc,
                                                    false, false);
      sacc = __builtin_amdgcn_wmma_i32_16x16x64_iu8(true, Aq[1], true, Bk[1], sacc,
                                                    false, false);
#pragma unroll
      for (int v = 0; v < 8; ++v) sf[nt][v] = (float)sacc[v] * sfac;
    }

#pragma unroll
    for (int v = 0; v < 8; ++v) {
      float tm = fmaxf(fmaxf(sf[0][v], sf[1][v]), fmaxf(sf[2][v], sf[3][v]));
      tm = fmaxf(tm, __shfl_xor(tm, 1, 32));
      tm = fmaxf(tm, __shfl_xor(tm, 2, 32));
      tm = fmaxf(tm, __shfl_xor(tm, 4, 32));
      tm = fmaxf(tm, __shfl_xor(tm, 8, 32));
      const float mnew = fmaxf(mrun[v], tm);
      const float fac = __expf(mrun[v] - mnew);
      mrun[v] = mnew;
      lrun[v] *= fac;
#pragma unroll
      for (int n = 0; n < 8; ++n) Oacc[n][v] = Oacc[n][v] * fac;
      float rs = 0.f;
#pragma unroll
      for (int nt = 0; nt < 4; ++nt) {
        const float p = __expf(sf[nt][v] - mnew);
        rs += p;
        myp[(v + 8 * hi) * 64 + nt * 16 + lane16] = (unsigned char)f32_to_e4m3(p * 448.f);
      }
      rs += __shfl_xor(rs, 1, 32);
      rs += __shfl_xor(rs, 2, 32);
      rs += __shfl_xor(rs, 4, 32);
      rs += __shfl_xor(rs, 8, 32);
      lrun[v] += rs;
    }

    asm volatile("s_wait_dscnt 0" ::: "memory");

    v8i Ap;
#pragma unroll
    for (int j = 0; j < 8; ++j)
      Ap[j] = *(const int*)(myp + lane16 * 64 + kbaseA8(j, hi));
#pragma unroll
    for (int n = 0; n < 8; ++n) {
      v8i Bv;
#pragma unroll
      for (int j = 0; j < 8; ++j)
        Bv[j] = *(const int*)(vcur + (n * 16 + lane16) * BLKK + kbaseB8(j, hi));
      Oacc[n] = __builtin_amdgcn_wmma_f32_16x16x64_fp8_fp8(Ap, Bv, (short)0, Oacc[n],
                                                           false, false);
    }
  }

#pragma unroll
  for (int n = 0; n < 8; ++n) {
    const int d = n * 16 + lane16;
    const float vs = vscale[bh * DD + d];
#pragma unroll
    for (int v = 0; v < 8; ++v) {
      const int l = qb * BLKQ + m0 + v + 8 * hi;
      const float denom = fmaxf(lrun[v], 1e-20f) * 448.f;
      const size_t oi = ((size_t)(b * LL + l) * HH + h) * DD + d;
      out[oi] = out[oi] + Oacc[n][v] * (vs / denom);
    }
  }
}

extern "C" void kernel_launch(void* const* d_in, const int* in_sizes, int n_in,
                              void* d_out, int out_size, void* d_ws, size_t ws_size,
                              hipStream_t stream) {
  (void)in_sizes; (void)n_in; (void)out_size; (void)ws_size;
  const float* Q  = (const float*)d_in[0];
  const float* K  = (const float*)d_in[1];
  const float* V  = (const float*)d_in[2];
  const float* PW = (const float*)d_in[3];
  const float* PB = (const float*)d_in[4];
  float* out = (float*)d_out;

  char* ws = (char*)d_ws;
  size_t off = 0;
  auto take = [&](size_t bytes) -> char* {
    char* p = ws + off;
    off += bytes;
    off = (off + 255) & ~(size_t)255;
    return p;
  };
  const size_t BH = (size_t)BB * HH;
  const size_t BHD = BH * DD;
  const size_t BHLD = BH * LL * DD;

  float*    km     = (float*)take(BHD * 4);
  float*    vscale = (float*)take(BHD * 4);
  float*    ksum   = (float*)take(BHD * 4);
  float*    kvf32  = (float*)take(BH * DD * DD * 4);
  float*    qp     = (float*)take(BH * NQB * DD * 4);
  float*    kp     = (float*)take(BH * NKB * DD * 4);
  float*    qscale = (float*)take(BH * NQB * 4);
  float*    kscale = (float*)take(BH * NKB * 4);
  unsigned* maskb  = (unsigned*)take(BH * NQB * 4);
  signed char*   q8  = (signed char*)take(BHLD);
  signed char*   k8  = (signed char*)take(BHLD);
  unsigned char* v8t = (unsigned char*)take(BHLD);
  __bf16* qfb  = (__bf16*)take(BHLD * 2);
  __bf16* kftb = (__bf16*)take(BHLD * 2);
  __bf16* vbt  = (__bf16*)take(BHLD * 2);
  __bf16* olb  = (__bf16*)take(BHLD * 2);
  __bf16* pwb  = (__bf16*)take((size_t)DD * DD * 2);

  k_prep<<<dim3(HH, BB), DD, 0, stream>>>(Q, K, V, km, vscale, ksum, kvf32, qp, kp);
  k_cvtw<<<(DD * DD + 255) / 256, 256, 0, stream>>>(PW, pwb);
  k_topk<<<dim3(HH, BB), 128, 0, stream>>>(qp, kp, maskb);
  k_quant_q<<<dim3(NQB, HH, BB), BLKQ, 0, stream>>>(Q, q8, qscale);
  k_quant_k<<<dim3(NKB, HH, BB), BLKK, 0, stream>>>(K, km, k8, kscale);
  k_rowwise<<<dim3(LL / 8, HH, BB), 256, 0, stream>>>(Q, K, V, vscale, ksum, qfb, kftb,
                                                      vbt, v8t);
  k_kvgemm<<<dim3(8, HH, BB), 256, 0, stream>>>(kftb, vbt, kvf32);
  k_lingemm<<<dim3(LL / 128, HH, BB), 128, 0, stream>>>(qfb, kvf32, ksum, olb);
  k_proj<<<dim3(LL / 128, HH, BB), 128, 0, stream>>>(olb, pwb, PB, out);
  k_attn<<<dim3(NQB, HH, BB), 256, 0, stream>>>(q8, k8, v8t, qscale, kscale, vscale,
                                                maskb, out);
}
